// MoELayer_54769422958703
// MI455X (gfx1250) — matrix operand order repaired and checked by reading, not compiled, not run on hardware
//
#include <hip/hip_runtime.h>
#include <hip/hip_bf16.h>
#include <math.h>

// ---------------------------------------------------------------------------
// MoE layer for MI455X (gfx1250): gather-based grouped GEMM with bf16 WMMA,
// async global->LDS staging (ASYNCcnt) and weight prefetch.
// ---------------------------------------------------------------------------

typedef __attribute__((ext_vector_type(16))) __bf16 v16bf;
typedef __attribute__((ext_vector_type(8)))  float  v8f;

#define N_TOK   8192
#define C_DIM   1024
#define E_NUM   8
#define H_DIM   2752
#define TOPKK   2
#define RSCALE  1.0f

#define BM 128
#define BN 64
#define BK 32

#define ROUTED_TILES 136                    // 2N/BM + E pad tiles
#define ROUTED_CAP   (ROUTED_TILES * BM)    // 17408
#define SHARED_OFF   ROUTED_CAP
#define SHARED_TILES (N_TOK / BM)           // 64
#define TOTAL_TILES  (ROUTED_TILES + SHARED_TILES)
#define TOTAL_ROWS   (ROUTED_CAP + N_TOK)   // 25600 rows of h

// workspace byte offsets (16B aligned where vector/async loads are used)
#define WS_COUNTS 0
#define WS_CURSOR 32
#define WS_OFF    64
#define WS_TIDX   128
#define WS_TW     (WS_TIDX + N_TOK * TOPKK * 4)   // 65664
#define WS_STOK   (WS_TW   + N_TOK * TOPKK * 4)   // 131200
#define WS_SW     (WS_STOK + ROUTED_CAP * 4)      // 200832
#define WS_H      (WS_SW   + ROUTED_CAP * 4)      // 270464 (16B aligned)
// total ws needed: WS_H + TOTAL_ROWS*H_DIM*2  ~= 141.2 MB

__device__ __forceinline__ unsigned short f2bf(float f) {
  unsigned u = __float_as_uint(f);
  unsigned lsb = (u >> 16) & 1u;
  u += 0x7FFFu + lsb;                 // round-to-nearest-even
  return (unsigned short)(u >> 16);
}

// pack two fp32 -> one dword of two bf16 (lo = first / even-k element)
__device__ __forceinline__ unsigned pack2(float lo, float hi) {
  return (unsigned)f2bf(lo) | ((unsigned)f2bf(hi) << 16);
}

union FragBF { v16bf v; unsigned u[8]; };
union AccF   { v8f v; float f[8]; };

__device__ __forceinline__ void zero_acc(AccF& a) {
#pragma unroll
  for (int i = 0; i < 8; ++i) a.f[i] = 0.f;
}

// A fragment: 16x32 bf16. lane m = lane%16; VGPR0-3: K = kh..kh+7,
// VGPR4-7: K = 16+kh..16+kh+7 where kh = 8*(lane>=16). As is row-major [BM][BK].
__device__ __forceinline__ FragBF load_a(const unsigned short* As, int mbase, int lane) {
  FragBF f;
  const unsigned short* r = As + (mbase + (lane & 15)) * BK;
  int kh = (lane >> 4) << 3;
#pragma unroll
  for (int i = 0; i < 4; ++i) f.u[i]     = *(const unsigned*)(r + kh + 2 * i);
#pragma unroll
  for (int i = 0; i < 4; ++i) f.u[4 + i] = *(const unsigned*)(r + 16 + kh + 2 * i);
  return f;
}

// B fragment: 32x16 bf16. lane col n = lane%16; VGPR0-3: K = kh..kh+7,
// VGPR4-7: K = 16+kh..16+kh+7 where kh = 8*(lane>=16). Bs is stored K-transposed: [BN][BK].
__device__ __forceinline__ FragBF load_b(const unsigned short* Bs, int nbase, int lane) {
  FragBF f;
  const unsigned short* p = Bs + (nbase + (lane & 15)) * BK + ((lane >> 4) << 3);
#pragma unroll
  for (int i = 0; i < 4; ++i) f.u[i] = *(const unsigned*)(p + 2 * i);
#pragma unroll
  for (int i = 0; i < 4; ++i) f.u[4 + i] = *(const unsigned*)(p + 16 + 2 * i);
  return f;
}

// gfx1250 async copy: 16B global -> LDS, tracked by ASYNCcnt
__device__ __forceinline__ void async_copy_b128(unsigned lds_off, const void* gaddr) {
  asm volatile("global_load_async_to_lds_b128 %0, %1, off"
               :: "v"(lds_off), "v"(gaddr) : "memory");
}
__device__ __forceinline__ void wait_async0() {
  asm volatile("s_wait_asynccnt 0" ::: "memory");
}

// ---------------------------------------------------------------------------
__global__ __launch_bounds__(256)
void init_kernel(float* __restrict__ y, int* __restrict__ stok, int* __restrict__ cc) {
  long i = (long)blockIdx.x * blockDim.x + threadIdx.x;
  if (i < (long)N_TOK * C_DIM) y[i] = 0.f;
  if (i < ROUTED_CAP) stok[i] = -1;
  if (i < 16) cc[i] = 0;              // counts[8] + cursor[8] are contiguous
}

// one wave per token: sigmoid scores, biased top-2, normalized weights
__global__ __launch_bounds__(256)
void gate_kernel(const float* __restrict__ x, const float* __restrict__ gw,
                 const float* __restrict__ ebias,
                 int* __restrict__ tidx, float* __restrict__ tw,
                 int* __restrict__ counts) {
  int wave = threadIdx.x >> 5, lane = threadIdx.x & 31;
  int t = blockIdx.x * 8 + wave;
  float acc[E_NUM];
#pragma unroll
  for (int e = 0; e < E_NUM; ++e) acc[e] = 0.f;
  const float* xt = x + (size_t)t * C_DIM;
  for (int k = lane; k < C_DIM; k += 32) {
    float xv = xt[k];
#pragma unroll
    for (int e = 0; e < E_NUM; ++e)
      acc[e] = fmaf(xv, gw[e * C_DIM + k], acc[e]);
  }
#pragma unroll
  for (int e = 0; e < E_NUM; ++e) {
#pragma unroll
    for (int off = 16; off; off >>= 1) acc[e] += __shfl_xor(acc[e], off, 32);
  }
  if (lane == 0) {
    float sc[E_NUM], rt[E_NUM];
#pragma unroll
    for (int e = 0; e < E_NUM; ++e) {
      sc[e] = 1.f / (1.f + __expf(-acc[e]));
      rt[e] = sc[e] + ebias[e];
    }
    int i0 = 0, i1 = -1;
    float r0 = rt[0], r1 = -1e30f;
#pragma unroll
    for (int e = 1; e < E_NUM; ++e) {
      float r = rt[e];
      if (r > r0) { r1 = r0; i1 = i0; r0 = r; i0 = e; }
      else if (r > r1) { r1 = r; i1 = e; }
    }
    float s0 = sc[i0], s1 = sc[i1];
    float inv = RSCALE / (s0 + s1 + 1e-8f);
    tidx[t * 2] = i0; tidx[t * 2 + 1] = i1;
    tw[t * 2] = s0 * inv; tw[t * 2 + 1] = s1 * inv;
    atomicAdd(&counts[i0], 1);
    atomicAdd(&counts[i1], 1);
  }
}

// per-expert slot offsets, padded to BM so GEMM tiles never straddle experts
__global__ void offsets_kernel(const int* __restrict__ counts, int* __restrict__ offs) {
  if (threadIdx.x == 0 && blockIdx.x == 0) {
    int o = 0;
#pragma unroll
    for (int e = 0; e < E_NUM; ++e) {
      offs[e] = o;
      o += (counts[e] + BM - 1) / BM * BM;
    }
    offs[E_NUM] = o;
  }
}

__global__ __launch_bounds__(256)
void scatter_kernel(const int* __restrict__ tidx, const float* __restrict__ tw,
                    const int* __restrict__ offs, int* __restrict__ cursor,
                    int* __restrict__ stok, float* __restrict__ sw) {
  int t = blockIdx.x * blockDim.x + threadIdx.x;
  if (t >= N_TOK) return;
#pragma unroll
  for (int k = 0; k < TOPKK; ++k) {
    int e = tidx[t * 2 + k];
    int pos = atomicAdd(&cursor[e], 1);
    int slot = offs[e] + pos;
    stok[slot] = t;
    sw[slot] = tw[t * 2 + k];
  }
}

// ---------------------------------------------------------------------------
// h[slot, :] = silu(x_g @ W1) * (x_g @ W3)   (bf16 output)
// grid: (TOTAL_TILES, H_DIM/BN), 256 threads = 8 waves, wave tile 32x32 per matrix
__global__ __launch_bounds__(256)
void moe_h_kernel(const float* __restrict__ x,
                  const float* __restrict__ w1, const float* __restrict__ w3,
                  const float* __restrict__ sw1, const float* __restrict__ sw3,
                  const int* __restrict__ stok, const int* __restrict__ offs,
                  unsigned short* __restrict__ hbuf) {
  __shared__ __align__(16) unsigned short As[BM * BK];
  __shared__ __align__(16) unsigned short B1s[BN * BK];
  __shared__ __align__(16) unsigned short B3s[BN * BK];

  const int tile = blockIdx.x;
  const int n0 = blockIdx.y * BN;
  const int tid = threadIdx.x;
  const int lane = tid & 31, wave = tid >> 5;
  const int wm = wave & 3, wn = wave >> 2;

  const float *W1, *W3;
  int slot_base;
  bool shared_ex;
  if (tile < ROUTED_TILES) {
    slot_base = tile * BM;
    if (slot_base >= offs[E_NUM]) return;     // uniform early-exit
    int e = 0;
    while (e < E_NUM - 1 && slot_base >= offs[e + 1]) ++e;
    W1 = w1 + (size_t)e * C_DIM * H_DIM;
    W3 = w3 + (size_t)e * C_DIM * H_DIM;
    shared_ex = false;
  } else {
    slot_base = SHARED_OFF + (tile - ROUTED_TILES) * BM;
    W1 = sw1; W3 = sw3;
    shared_ex = true;
  }

  const int ar = tid >> 1;              // A-stage row 0..127
  const int ah = (tid & 1) * 16;        // 16-float half
  int tok = shared_ex ? (slot_base - SHARED_OFF + ar) : stok[slot_base + ar];

  const int kp = tid & 15;              // B-stage K-pair: k = 2*kp, 2*kp+1
  const int ng = (tid >> 4) * 4;        // B-stage first of 4 consecutive n

  AccF a1acc[2][2], a3acc[2][2];
#pragma unroll
  for (int i = 0; i < 2; ++i)
#pragma unroll
    for (int j = 0; j < 2; ++j) { zero_acc(a1acc[i][j]); zero_acc(a3acc[i][j]); }

  for (int k0 = 0; k0 < C_DIM; k0 += BK) {
    // stage A (gathered x rows, fp32 -> bf16, packed dword LDS stores)
    {
      float4 v0, v1, v2, v3;
      if (tok >= 0) {
        const float4* s = (const float4*)(x + (size_t)tok * C_DIM + k0 + ah);
        v0 = s[0]; v1 = s[1]; v2 = s[2]; v3 = s[3];
      } else {
        v0 = v1 = v2 = v3 = make_float4(0.f, 0.f, 0.f, 0.f);
      }
      unsigned* d = (unsigned*)(As + ar * BK + ah);
      d[0] = pack2(v0.x, v0.y); d[1] = pack2(v0.z, v0.w);
      d[2] = pack2(v1.x, v1.y); d[3] = pack2(v1.z, v1.w);
      d[4] = pack2(v2.x, v2.y); d[5] = pack2(v2.z, v2.w);
      d[6] = pack2(v3.x, v3.y); d[7] = pack2(v3.z, v3.w);
    }
    // stage B1/B3: two K rows per thread, packed (k, k+1) dword per n
    {
      const float* r1 = W1 + (size_t)(k0 + 2 * kp) * H_DIM + n0 + ng;
      const float* r3 = W3 + (size_t)(k0 + 2 * kp) * H_DIM + n0 + ng;
      if (k0 + BK < C_DIM) {                 // prefetch next K-panel
        __builtin_prefetch(r1 + (size_t)BK * H_DIM, 0, 0);
        __builtin_prefetch(r3 + (size_t)BK * H_DIM, 0, 0);
      }
      float4 e0 = *(const float4*)r1;
      float4 e1 = *(const float4*)(r1 + H_DIM);
      unsigned* B1p = (unsigned*)B1s;
      B1p[(ng + 0) * (BK / 2) + kp] = pack2(e0.x, e1.x);
      B1p[(ng + 1) * (BK / 2) + kp] = pack2(e0.y, e1.y);
      B1p[(ng + 2) * (BK / 2) + kp] = pack2(e0.z, e1.z);
      B1p[(ng + 3) * (BK / 2) + kp] = pack2(e0.w, e1.w);
      e0 = *(const float4*)r3;
      e1 = *(const float4*)(r3 + H_DIM);
      unsigned* B3p = (unsigned*)B3s;
      B3p[(ng + 0) * (BK / 2) + kp] = pack2(e0.x, e1.x);
      B3p[(ng + 1) * (BK / 2) + kp] = pack2(e0.y, e1.y);
      B3p[(ng + 2) * (BK / 2) + kp] = pack2(e0.z, e1.z);
      B3p[(ng + 3) * (BK / 2) + kp] = pack2(e0.w, e1.w);
    }
    __syncthreads();

    // hoist all fragment loads so WMMAs can overlap the DS traffic
    FragBF a0  = load_a(As, wm * 32, lane);
    FragBF a1  = load_a(As, wm * 32 + 16, lane);
    FragBF b10 = load_b(B1s, wn * 32, lane);
    FragBF b11 = load_b(B1s, wn * 32 + 16, lane);
    FragBF b30 = load_b(B3s, wn * 32, lane);
    FragBF b31 = load_b(B3s, wn * 32 + 16, lane);

    a1acc[0][0].v = __builtin_amdgcn_wmma_f32_16x16x32_bf16(false, a0.v, false, b10.v, (short)0, a1acc[0][0].v, false, false);
    a1acc[1][0].v = __builtin_amdgcn_wmma_f32_16x16x32_bf16(false, a1.v, false, b10.v, (short)0, a1acc[1][0].v, false, false);
    a1acc[0][1].v = __builtin_amdgcn_wmma_f32_16x16x32_bf16(false, a0.v, false, b11.v, (short)0, a1acc[0][1].v, false, false);
    a1acc[1][1].v = __builtin_amdgcn_wmma_f32_16x16x32_bf16(false, a1.v, false, b11.v, (short)0, a1acc[1][1].v, false, false);
    a3acc[0][0].v = __builtin_amdgcn_wmma_f32_16x16x32_bf16(false, a0.v, false, b30.v, (short)0, a3acc[0][0].v, false, false);
    a3acc[1][0].v = __builtin_amdgcn_wmma_f32_16x16x32_bf16(false, a1.v, false, b30.v, (short)0, a3acc[1][0].v, false, false);
    a3acc[0][1].v = __builtin_amdgcn_wmma_f32_16x16x32_bf16(false, a0.v, false, b31.v, (short)0, a3acc[0][1].v, false, false);
    a3acc[1][1].v = __builtin_amdgcn_wmma_f32_16x16x32_bf16(false, a1.v, false, b31.v, (short)0, a3acc[1][1].v, false, false);
    __syncthreads();
  }

  // epilogue: h = silu(g) * u, bf16 store
  const int col = lane & 15;
  const int rhl = (lane >> 4) * 8;
#pragma unroll
  for (int ms = 0; ms < 2; ++ms)
#pragma unroll
    for (int ns = 0; ns < 2; ++ns) {
      int ngl = n0 + wn * 32 + ns * 16 + col;
#pragma unroll
      for (int v = 0; v < 8; ++v) {
        int m = wm * 32 + ms * 16 + rhl + v;
        float g = a1acc[ms][ns].f[v];
        float s = g / (1.f + __expf(-g));
        float hv = s * a3acc[ms][ns].f[v];
        hbuf[(size_t)(slot_base + m) * H_DIM + ngl] = f2bf(hv);
      }
    }
}

// ---------------------------------------------------------------------------
// y[token, :] += gate_w[slot] * (h[slot, :] @ W2)    (atomic scatter combine)
// A-tile is already bf16 in global: staged with async global->LDS (ASYNCcnt).
// grid: (TOTAL_TILES, C_DIM/BN)
__global__ __launch_bounds__(256)
void moe_out_kernel(const unsigned short* __restrict__ hbuf,
                    const float* __restrict__ w2, const float* __restrict__ sw2,
                    const int* __restrict__ stok, const float* __restrict__ sw,
                    const int* __restrict__ offs, float* __restrict__ y) {
  __shared__ __align__(16) unsigned short As[BM * BK];
  __shared__ __align__(16) unsigned short Bs[BN * BK];

  const int tile = blockIdx.x;
  const int n0 = blockIdx.y * BN;
  const int tid = threadIdx.x;
  const int lane = tid & 31, wave = tid >> 5;
  const int wm = wave & 3, wn = wave >> 2;

  const float* W2;
  int slot_base;
  bool shared_ex;
  if (tile < ROUTED_TILES) {
    slot_base = tile * BM;
    if (slot_base >= offs[E_NUM]) return;
    int e = 0;
    while (e < E_NUM - 1 && slot_base >= offs[e + 1]) ++e;
    W2 = w2 + (size_t)e * H_DIM * C_DIM;
    shared_ex = false;
  } else {
    slot_base = SHARED_OFF + (tile - ROUTED_TILES) * BM;
    W2 = sw2;
    shared_ex = true;
  }

  const int ar = tid >> 1;            // A-stage row
  const int ah = (tid & 1) * 16;      // 16-element (bf16) half
  const int kp = tid & 15;            // B-stage K-pair
  const int ng = (tid >> 4) * 4;      // B-stage first of 4 consecutive n

  const unsigned lds_dst = (unsigned)(uintptr_t)(As + ar * BK + ah);
  const unsigned short* hrow = hbuf + (size_t)(slot_base + ar) * H_DIM + ah;

  AccF acc[2][2];
#pragma unroll
  for (int i = 0; i < 2; ++i)
#pragma unroll
    for (int j = 0; j < 2; ++j) zero_acc(acc[i][j]);

  for (int k0 = 0; k0 < H_DIM; k0 += BK) {
    // stage A: async DMA 32B per thread, no VGPR round-trip
    async_copy_b128(lds_dst, hrow + k0);
    async_copy_b128(lds_dst + 16, hrow + k0 + 8);
    // stage B: W2 is (H, C) row-major, two K rows per thread, packed dwords
    {
      const float* r = W2 + (size_t)(k0 + 2 * kp) * C_DIM + n0 + ng;
      if (k0 + BK < H_DIM)
        __builtin_prefetch(r + (size_t)BK * C_DIM, 0, 0);
      float4 e0 = *(const float4*)r;
      float4 e1 = *(const float4*)(r + C_DIM);
      unsigned* Bp = (unsigned*)Bs;
      Bp[(ng + 0) * (BK / 2) + kp] = pack2(e0.x, e1.x);
      Bp[(ng + 1) * (BK / 2) + kp] = pack2(e0.y, e1.y);
      Bp[(ng + 2) * (BK / 2) + kp] = pack2(e0.z, e1.z);
      Bp[(ng + 3) * (BK / 2) + kp] = pack2(e0.w, e1.w);
    }
    wait_async0();                     // A-tile landed in LDS
    __syncthreads();

    FragBF a0 = load_a(As, wm * 32, lane);
    FragBF a1 = load_a(As, wm * 32 + 16, lane);
    FragBF b0 = load_b(Bs, wn * 32, lane);
    FragBF b1 = load_b(Bs, wn * 32 + 16, lane);
    acc[0][0].v = __builtin_amdgcn_wmma_f32_16x16x32_bf16(false, a0.v, false, b0.v, (short)0, acc[0][0].v, false, false);
    acc[1][0].v = __builtin_amdgcn_wmma_f32_16x16x32_bf16(false, a1.v, false, b0.v, (short)0, acc[1][0].v, false, false);
    acc[0][1].v = __builtin_amdgcn_wmma_f32_16x16x32_bf16(false, a0.v, false, b1.v, (short)0, acc[0][1].v, false, false);
    acc[1][1].v = __builtin_amdgcn_wmma_f32_16x16x32_bf16(false, a1.v, false, b1.v, (short)0, acc[1][1].v, false, false);
    __syncthreads();
  }

  const int col = lane & 15;
  const int rhl = (lane >> 4) * 8;
#pragma unroll
  for (int ms = 0; ms < 2; ++ms) {
#pragma unroll
    for (int v = 0; v < 8; ++v) {
      int m = wm * 32 + ms * 16 + rhl + v;
      int slot = slot_base + m;
      int tok = shared_ex ? (slot - SHARED_OFF) : stok[slot];
      float sc = shared_ex ? 1.f : sw[slot];
      if (tok >= 0) {
#pragma unroll
        for (int ns = 0; ns < 2; ++ns) {
          int ngl = n0 + wn * 32 + ns * 16 + col;
          atomicAdd(&y[(size_t)tok * C_DIM + ngl], acc[ms][ns].f[v] * sc);
        }
      }
    }
  }
}

// ---------------------------------------------------------------------------
extern "C" void kernel_launch(void* const* d_in, const int* in_sizes, int n_in,
                              void* d_out, int out_size, void* d_ws, size_t ws_size,
                              hipStream_t stream) {
  (void)in_sizes; (void)n_in; (void)out_size; (void)ws_size;
  const float* x   = (const float*)d_in[0];
  const float* gw  = (const float*)d_in[1];
  const float* eb  = (const float*)d_in[2];
  const float* w1  = (const float*)d_in[3];
  const float* w2  = (const float*)d_in[4];
  const float* w3  = (const float*)d_in[5];
  const float* sw1 = (const float*)d_in[6];
  const float* sw2 = (const float*)d_in[7];
  const float* sw3 = (const float*)d_in[8];
  float* y = (float*)d_out;

  char* ws = (char*)d_ws;
  int* counts = (int*)(ws + WS_COUNTS);
  int* cursor = (int*)(ws + WS_CURSOR);
  int* offs   = (int*)(ws + WS_OFF);
  int* tidx   = (int*)(ws + WS_TIDX);
  float* tw   = (float*)(ws + WS_TW);
  int* stok   = (int*)(ws + WS_STOK);
  float* swt  = (float*)(ws + WS_SW);
  unsigned short* hbuf = (unsigned short*)(ws + WS_H);

  init_kernel<<<(N_TOK * C_DIM + 255) / 256, 256, 0, stream>>>(y, stok, counts);
  gate_kernel<<<N_TOK / 8, 256, 0, stream>>>(x, gw, eb, tidx, tw, counts);
  offsets_kernel<<<1, 32, 0, stream>>>(counts, offs);
  scatter_kernel<<<N_TOK / 256, 256, 0, stream>>>(tidx, tw, offs, cursor, stok, swt);

  dim3 g1(TOTAL_TILES, H_DIM / BN);
  moe_h_kernel<<<g1, 256, 0, stream>>>(x, w1, w3, sw1, sw3, stok, offs, hbuf);

  dim3 g2(TOTAL_TILES, C_DIM / BN);
  moe_out_kernel<<<g2, 256, 0, stream>>>(hbuf, w2, sw2, stok, swt, offs, y);
}
